// FullModel_2826088481010
// MI455X (gfx1250) — matrix operand order repaired and checked by reading, not compiled, not run on hardware
//
#include <hip/hip_runtime.h>
#include <stdint.h>

typedef __attribute__((ext_vector_type(16))) _Float16 v16h;
typedef __attribute__((ext_vector_type(8)))  _Float16 v8h;
typedef __attribute__((ext_vector_type(8)))  float    v8f;

#define S_PTS   32768
#define P_TOT   (50*240*480)
#define DEV_AMT 0.01f

// ---- d_out float offsets (reference return-tuple order, flat) ----
#define O_RGBCUR 0
#define O_ALPHAC 98304
#define O_XYZP   131072
#define O_UVWF   917504
#define O_UVWB   1015808
#define O_UVWPF  1114112
#define O_UVWPB  1900544
#define O_ALPHA  2686976
#define O_MRGBF  2719744
#define O_MRGBB  2818048
#define O_MRGB   2916352
#define O_OCC    3014656

// ---- workspace byte offsets ----
#define WS_GSUM   0
#define WS_PART   16
#define WS_WTS    2048
#define WS_XYZT   720896
#define WS_XYZTP  1245184

// ---- f16 weight pack layout (per network, half-element offsets) ----
#define NET_STRIDE_H 68608
#define OFF_W1   0        // [K][256] (K = IN)
#define OFF_B1   1024
#define OFF_W2T  1280     // transposed [N=256][K=256]
#define OFF_B2   66816
#define OFF_W3T  67072    // transposed [N=OUT][K=256]
#define OFF_B3   67840

struct WPtrs { const float* p[30]; };

// =====================================================================
// Weight conversion: fp32 -> f16, W2/W3 transposed for B-fragment loads
// =====================================================================
__global__ void convert_weights_kernel(WPtrs wp, _Float16* wts) {
  const int net = blockIdx.x;
  const int tid = threadIdx.x;
  const int in_dims[5]  = {4, 4, 4, 3, 3};
  const int out_dims[5] = {3, 3, 1, 3, 3};
  const int IN  = in_dims[net];
  const int OUT = out_dims[net];
  const float* W1 = wp.p[net*6 + 0];
  const float* B1 = wp.p[net*6 + 1];
  const float* W2 = wp.p[net*6 + 2];
  const float* B2 = wp.p[net*6 + 3];
  const float* W3 = wp.p[net*6 + 4];
  const float* B3 = wp.p[net*6 + 5];
  _Float16* dst = wts + (size_t)net * NET_STRIDE_H;

  for (int idx = tid; idx < IN*256; idx += 256) dst[OFF_W1 + idx] = (_Float16)W1[idx];
  dst[OFF_B1 + tid] = (_Float16)B1[tid];
  for (int r = 0; r < 256; ++r) {              // W2T[n][k] = W2[k][n]
    int idx = r*256 + tid;
    int n = idx >> 8, k = idx & 255;
    dst[OFF_W2T + n*256 + k] = (_Float16)W2[k*256 + n];
  }
  dst[OFF_B2 + tid] = (_Float16)B2[tid];
  for (int idx = tid; idx < OUT*256; idx += 256) {   // W3T[n][k] = W3[k][n]
    int n = idx >> 8, k = idx & 255;
    dst[OFF_W3T + idx] = (_Float16)W3[k*OUT + n];
  }
  if (tid < OUT) dst[OFF_B3 + tid] = (_Float16)B3[tid];
}

// =====================================================================
// Prep: gathers, xyzt / xyzt_p construction, diff + block-partial sums
// =====================================================================
__global__ void prep_kernel(const float* __restrict__ gt_rgb,
                            const float* __restrict__ gt_mask,
                            const float* __restrict__ xyzt_all,
                            const int*   __restrict__ xyt_all,
                            const int*   __restrict__ inds,
                            float* __restrict__ out,
                            float* __restrict__ ws_xyzt,
                            float* __restrict__ ws_xyztp,
                            float* __restrict__ partials) {
  const int i   = blockIdx.x * 256 + threadIdx.x;
  const int idx = inds[i];
  const int xi = xyt_all[idx];
  const int yi = xyt_all[P_TOT + idx];
  const int ti = xyt_all[2*P_TOT + idx];
#pragma unroll
  for (int c = 0; c < 3; ++c)
    out[O_RGBCUR + i*3 + c] = gt_rgb[((ti*3 + c)*240 + yi)*480 + xi];
  out[O_ALPHAC + i] = gt_mask[(ti*240 + yi)*480 + xi];

  const float x  = xyzt_all[idx];
  const float y  = xyzt_all[P_TOT + idx];
  const float z  = xyzt_all[2*P_TOT + idx];
  const float tn = xyzt_all[3*P_TOT + idx] * (1.0f/25.0f) - 1.0f;
  ws_xyzt[i*4+0] = x; ws_xyzt[i*4+1] = y; ws_xyzt[i*4+2] = z; ws_xyzt[i*4+3] = tn;

  float s0 = 0.f, s1 = 0.f, s2 = 0.f;
#pragma unroll
  for (int j = 0; j < 8; ++j) {
    float ax = x + ((j & 4) ? -DEV_AMT : DEV_AMT);
    float ay = y + ((j & 2) ? -DEV_AMT : DEV_AMT);
    float az = z + ((j & 1) ? -DEV_AMT : DEV_AMT);
    float inv = 1.0f / sqrtf(ax*ax + ay*ay + az*az);
    ax *= inv; ay *= inv; az *= inv;
    const int r = (i*8 + j)*4;
    ws_xyztp[r+0] = ax; ws_xyztp[r+1] = ay; ws_xyztp[r+2] = az; ws_xyztp[r+3] = tn;
    const float dx = x - ax, dy = y - ay, dz = z - az;
    const int d = (i*8 + j)*3;
    out[O_XYZP + d + 0] = dx; out[O_XYZP + d + 1] = dy; out[O_XYZP + d + 2] = dz;
    s0 += dx*dx; s1 += dy*dy; s2 += dz*dz;
  }
  __shared__ float bs[3];
  if (threadIdx.x < 3) bs[threadIdx.x] = 0.f;
  __syncthreads();
  atomicAdd(&bs[0], s0); atomicAdd(&bs[1], s1); atomicAdd(&bs[2], s2);
  __syncthreads();
  if (threadIdx.x < 3) partials[blockIdx.x*3 + threadIdx.x] = bs[threadIdx.x];
}

__global__ void reduce_gnorm_kernel(const float* __restrict__ partials,
                                    float* __restrict__ gnorm) {
  if (threadIdx.x < 3) {
    float s = 0.f;
    for (int b = 0; b < 128; ++b) s += partials[b*3 + threadIdx.x];
    gnorm[threadIdx.x] = sqrtf(s);
  }
}

__global__ void xyzp_scale_kernel(float* __restrict__ out,
                                  const float* __restrict__ gnorm) {
  const int i = blockIdx.x * 256 + threadIdx.x;      // < S*8*3
  const int c = i - (i/3)*3;
  out[O_XYZP + i] = out[O_XYZP + i] / gnorm[c];
}

__global__ void combine_kernel(float* __restrict__ out) {
  const int i = blockIdx.x * 256 + threadIdx.x;      // < S*3
  const float a = out[O_ALPHA + i/3];
  const float f = out[O_MRGBF + i];
  const float b = out[O_MRGBB + i];
  out[O_MRGB + i] = f*a + b*(1.0f - a);
  out[O_OCC  + i] = f*(1.0f - a);
}

// =====================================================================
// Fused MLP: [IN->256] ReLU -> [256->256] ReLU (WMMA) -> [256->OUT] (WMMA)
//   256 threads = 8 waves, 128 points per block, weights async-staged in LDS
// =====================================================================
template<int IN, int OUT, int ACT>  // ACT: 0=tanh, 1=sigmoid
__global__ __launch_bounds__(256, 1)
void mlp_kernel(const float* __restrict__ x,
                const _Float16* __restrict__ w,
                float* __restrict__ out) {
  constexpr int BM    = 128;
  constexpr int PITCH = 264;   // halves; 528B rows -> 16B aligned, bank-staggered

  __shared__ __align__(16) _Float16 lds_h1[BM * PITCH];
  __shared__ __align__(16) _Float16 lds_h2[BM * PITCH];
  __shared__ __align__(16) _Float16 lds_w [256 * PITCH];

  const int tid  = threadIdx.x;
  const int lane = tid & 31;
  const int wave = tid >> 5;
  const int m0g  = blockIdx.x * BM;

  const _Float16* gw1  = w + OFF_W1;
  const _Float16* gb1  = w + OFF_B1;
  const _Float16* gw2t = w + OFF_W2T;
  const _Float16* gb2  = w + OFF_B2;
  const _Float16* gw3t = w + OFF_W3T;
  const _Float16* gb3  = w + OFF_B3;

  // ---- async-copy W2^T (256x256 f16) into pitch-padded LDS (TDM-class path)
  {
    const uint32_t lw = (uint32_t)(uintptr_t)(&lds_w[0]);
#pragma unroll 4
    for (int r = 0; r < 32; ++r) {
      const int idx = (r << 8) + tid;     // 8192 chunks of 16B
      const int row = idx >> 5;           // 0..255
      const int col = idx & 31;           // 16B chunk within row
      uint64_t ga = (uint64_t)(uintptr_t)gw2t + (uint64_t)(row*512 + col*16);
      uint32_t la = lw + (uint32_t)(row*(PITCH*2) + col*16);
      asm volatile("global_load_async_to_lds_b128 %0, %1, off"
                   :: "v"(la), "v"(ga) : "memory");
    }
  }

  // ---- layer 1 (K = IN, cheap VALU), overlapped with async weight copy ----
  {
    const int m  = tid >> 1;
    const int nb = (tid & 1) * 128;
    float xi[IN];
#pragma unroll
    for (int k = 0; k < IN; ++k) xi[k] = x[(m0g + m)*IN + k];
    for (int nn = 0; nn < 128; ++nn) {
      const int n = nb + nn;
      float acc = (float)gb1[n];
#pragma unroll
      for (int k = 0; k < IN; ++k) acc += xi[k] * (float)gw1[k*256 + n];
      lds_h1[m*PITCH + n] = (_Float16)fmaxf(acc, 0.0f);
    }
  }

  asm volatile("s_wait_asynccnt 0" ::: "memory");
  __syncthreads();

  const int mrow = wave * 16;
  const int ksel = lane >> 4;       // 0: K 0-7,16-23; 1: K 8-15,24-31 of each 32 (ISA layout)
  const int nlan = lane & 15;

  // ---- layer 2: per wave 16x256, K=256 in 8 chunks, N in 4 chunks of 64 ----
  for (int nc = 0; nc < 4; ++nc) {
    v8f acc[4];
#pragma unroll
    for (int t = 0; t < 4; ++t) {
      const float b = (float)gb2[nc*64 + t*16 + nlan];
#pragma unroll
      for (int r = 0; r < 8; ++r) acc[t][r] = b;
    }
    for (int kc = 0; kc < 8; ++kc) {
      const int kb = kc * 32;
      union { v16h v; v8h h[2]; } A;
      A.h[0] = *(const v8h*)&lds_h1[(mrow + nlan)*PITCH + kb + ksel*8];
      A.h[1] = *(const v8h*)&lds_h1[(mrow + nlan)*PITCH + kb + 16 + ksel*8];
#pragma unroll
      for (int t = 0; t < 4; ++t) {
        const int n = nc*64 + t*16 + nlan;
        union { v16h v; v8h h[2]; } B;
        B.h[0] = *(const v8h*)&lds_w[n*PITCH + kb + ksel*8];
        B.h[1] = *(const v8h*)&lds_w[n*PITCH + kb + 16 + ksel*8];
        acc[t] = __builtin_amdgcn_wmma_f32_16x16x32_f16(
            false, A.v, false, B.v, (short)0, acc[t], false, false);
      }
    }
#pragma unroll
    for (int t = 0; t < 4; ++t) {
      const int n = nc*64 + t*16 + nlan;
#pragma unroll
      for (int r = 0; r < 8; ++r) {
        const int m = mrow + ksel*8 + r;
        lds_h2[m*PITCH + n] = (_Float16)fmaxf(acc[t][r], 0.0f);
      }
    }
  }

  // ---- layer 3: one 16-wide N tile (only OUT columns valid) ----
  {
    v8f acc;
    const float b3v = (nlan < OUT) ? (float)gb3[nlan] : 0.0f;
#pragma unroll
    for (int r = 0; r < 8; ++r) acc[r] = b3v;
    const v8h zero8 = {};
    for (int kc = 0; kc < 8; ++kc) {
      const int kb = kc * 32;
      union { v16h v; v8h h[2]; } A;
      A.h[0] = *(const v8h*)&lds_h2[(mrow + nlan)*PITCH + kb + ksel*8];
      A.h[1] = *(const v8h*)&lds_h2[(mrow + nlan)*PITCH + kb + 16 + ksel*8];
      union { v16h v; v8h h[2]; } B;
      B.h[0] = zero8; B.h[1] = zero8;
      if (nlan < OUT) {
        const _Float16* p = gw3t + nlan*256 + kb + ksel*8;
        B.h[0] = *(const v8h*)p;
        B.h[1] = *(const v8h*)(p + 16);
      }
      acc = __builtin_amdgcn_wmma_f32_16x16x32_f16(
          false, A.v, false, B.v, (short)0, acc, false, false);
    }
    if (nlan < OUT) {
#pragma unroll
      for (int r = 0; r < 8; ++r) {
        const int m = m0g + mrow + ksel*8 + r;
        const float v = acc[r];
        out[m*OUT + nlan] = (ACT == 0) ? tanhf(v) : (1.0f / (1.0f + expf(-v)));
      }
    }
  }
}

// =====================================================================
extern "C" void kernel_launch(void* const* d_in, const int* in_sizes, int n_in,
                              void* d_out, int out_size, void* d_ws, size_t ws_size,
                              hipStream_t stream) {
  (void)in_sizes; (void)n_in; (void)out_size; (void)ws_size;
  const float* gt_rgb   = (const float*)d_in[0];
  const float* gt_mask  = (const float*)d_in[1];
  const float* xyzt_all = (const float*)d_in[2];
  const int*   xyt_all  = (const int*)d_in[3];
  const int*   inds     = (const int*)d_in[4];
  WPtrs wp;
  for (int i = 0; i < 30; ++i) wp.p[i] = (const float*)d_in[5 + i];

  char* ws = (char*)d_ws;
  float*    gnorm = (float*)(ws + WS_GSUM);
  float*    part  = (float*)(ws + WS_PART);
  _Float16* wts   = (_Float16*)(ws + WS_WTS);
  float*    xyzt  = (float*)(ws + WS_XYZT);
  float*    xyztp = (float*)(ws + WS_XYZTP);
  float*    out   = (float*)d_out;

  convert_weights_kernel<<<5, 256, 0, stream>>>(wp, wts);
  prep_kernel<<<128, 256, 0, stream>>>(gt_rgb, gt_mask, xyzt_all, xyt_all, inds,
                                       out, xyzt, xyztp, part);
  reduce_gnorm_kernel<<<1, 32, 0, stream>>>(part, gnorm);

  // nets: 0=pc_f 1=pc_b 2=pa 3=pr_f 4=pr_b
  mlp_kernel<4,3,0><<< 256, 256, 0, stream>>>(xyzt,  wts + 0*NET_STRIDE_H, out + O_UVWF);
  mlp_kernel<4,3,0><<< 256, 256, 0, stream>>>(xyzt,  wts + 1*NET_STRIDE_H, out + O_UVWB);
  mlp_kernel<4,3,0><<<2048, 256, 0, stream>>>(xyztp, wts + 0*NET_STRIDE_H, out + O_UVWPF);
  mlp_kernel<4,3,0><<<2048, 256, 0, stream>>>(xyztp, wts + 1*NET_STRIDE_H, out + O_UVWPB);
  mlp_kernel<4,1,1><<< 256, 256, 0, stream>>>(xyzt,  wts + 2*NET_STRIDE_H, out + O_ALPHA);
  mlp_kernel<3,3,1><<< 256, 256, 0, stream>>>(out + O_UVWF, wts + 3*NET_STRIDE_H, out + O_MRGBF);
  mlp_kernel<3,3,1><<< 256, 256, 0, stream>>>(out + O_UVWB, wts + 4*NET_STRIDE_H, out + O_MRGBB);

  xyzp_scale_kernel<<<3072, 256, 0, stream>>>(out, gnorm);
  combine_kernel<<< 384, 256, 0, stream>>>(out);
}
